// Flow_44220983280312
// MI455X (gfx1250) — matrix operand order repaired and checked by reading, not compiled, not run on hardware
//
#include <hip/hip_runtime.h>
#include <math.h>

typedef __attribute__((ext_vector_type(16))) __bf16 v16bf;
typedef __attribute__((ext_vector_type(8)))  float  v8f;

#define WAVES_PER_BLOCK 4
#define TILES_PER_WAVE  4
// B = 32768 batches, one 16-row M-tile per batch
#define NUM_TILES 32768

__device__ __forceinline__ float softplusf(float v) {
    return v > 15.f ? v : __logf(1.f + __expf(v));
}

__global__ __launch_bounds__(128) void flow_fused_kernel(
    const float* __restrict__ x,  const float* __restrict__ c,
    const float* __restrict__ W1, const float* __restrict__ b1,
    const float* __restrict__ W2, const float* __restrict__ b2,
    float* __restrict__ out)
{
    // W1 fragments: 16 ntiles x 4 ksteps, 32 lanes x 16 bf16 each  = 64 KB
    __shared__ alignas(32) __bf16 sW1[64 * 512];
    // W2 fragments: 7 ntiles x 8 ksteps (N padded 97->112)         = 56 KB
    __shared__ alignas(32) __bf16 sW2[56 * 512];
    // per-wave tile buffer: h (16x256 bf16 = 8KB) reused as theta (16x112 f32 = 7KB)
    __shared__ alignas(32) char   sBuf[WAVES_PER_BLOCK][8192];

    const int tid   = threadIdx.x;
    const int l     = tid & 31;
    const int wid   = tid >> 5;
    const int row   = l & 15;       // A-layout / C-layout row & col id
    const int khalf = l >> 4;
    const int kb    = khalf * 8;    // A-fragment k base; also C/D row base

    // ---- stage W1 as bf16 WMMA B-fragments (B is 32x16, lane=col, khalf picks K-half) ----
    for (int f = wid; f < 64; f += WAVES_PER_BLOCK) {
        const int nt = f >> 2, ks = f & 3;
        const int col = nt * 16 + row;
        const int k0  = ks * 32 + khalf * 8;
        __bf16* dst = &sW1[f * 512 + l * 16];
        #pragma unroll
        for (int i = 0; i < 16; ++i)
            dst[i] = (__bf16)W1[(size_t)(k0 + i + (i >= 8 ? 8 : 0)) * 256 + col];
    }
    // ---- stage W2 (zero-pad cols 97..111) ----
    for (int f = wid; f < 56; f += WAVES_PER_BLOCK) {
        const int nt = f >> 3, ks = f & 7;
        const int col = nt * 16 + row;
        const int k0  = ks * 32 + khalf * 8;
        __bf16* dst = &sW2[f * 512 + l * 16];
        #pragma unroll
        for (int i = 0; i < 16; ++i)
            dst[i] = (col < 97) ? (__bf16)W2[(size_t)(k0 + i + (i >= 8 ? 8 : 0)) * 97 + col] : (__bf16)0.f;
    }
    __syncthreads();

    __bf16* hb = (__bf16*)&sBuf[wid][0];
    float*  tb = (float*) &sBuf[wid][0];

    // ---- hoist per-lane bias values (loop-invariant across tiles) ----
    float b1r[16];
    #pragma unroll
    for (int nt = 0; nt < 16; ++nt) b1r[nt] = b1[nt * 16 + row];
    float b2r[7];
    #pragma unroll
    for (int nt = 0; nt < 7; ++nt) {
        const int cb = nt * 16 + row;
        b2r[nt] = (cb < 97) ? b2[cb] : 0.f;
    }

    const int gwave = blockIdx.x * WAVES_PER_BLOCK + wid;

    for (int j = 0; j < TILES_PER_WAVE; ++j) {
        const int    t  = gwave * TILES_PER_WAVE + j;
        const size_t r0 = (size_t)t * 16;
        const float* crow = c + (r0 + row) * 128;

        if (j + 1 < TILES_PER_WAVE) {
            const char* nxt = (const char*)(c + (r0 + 16) * 128);
            __builtin_prefetch(nxt + l * 256, 0, 0);
            __builtin_prefetch(nxt + l * 256 + 128, 0, 0);
        }

        // ---- A fragments for GEMM1: 16x32 bf16 per K-step (f32 load, cvt) ----
        v16bf a1[4];
        #pragma unroll
        for (int ks = 0; ks < 4; ++ks) {
            const float4 q0 = *reinterpret_cast<const float4*>(crow + ks * 32 + kb);
            const float4 q1 = *reinterpret_cast<const float4*>(crow + ks * 32 + kb + 4);
            const float4 q2 = *reinterpret_cast<const float4*>(crow + ks * 32 + kb + 16);
            const float4 q3 = *reinterpret_cast<const float4*>(crow + ks * 32 + kb + 20);
            v16bf a;
            a[0] = (__bf16)q0.x;  a[1] = (__bf16)q0.y;  a[2] = (__bf16)q0.z;  a[3] = (__bf16)q0.w;
            a[4] = (__bf16)q1.x;  a[5] = (__bf16)q1.y;  a[6] = (__bf16)q1.z;  a[7] = (__bf16)q1.w;
            a[8] = (__bf16)q2.x;  a[9] = (__bf16)q2.y;  a[10] = (__bf16)q2.z; a[11] = (__bf16)q2.w;
            a[12] = (__bf16)q3.x; a[13] = (__bf16)q3.y; a[14] = (__bf16)q3.z; a[15] = (__bf16)q3.w;
            a1[ks] = a;
        }

        // ---- GEMM1: h = relu(c @ W1 + b1), 16x256, store bf16 tile to LDS ----
        #pragma unroll
        for (int nt = 0; nt < 16; ++nt) {
            v8f acc = {};
            #pragma unroll
            for (int ks = 0; ks < 4; ++ks) {
                v16bf b = *(const v16bf*)&sW1[(nt * 4 + ks) * 512 + l * 16];
                acc = __builtin_amdgcn_wmma_f32_16x16x32_bf16(false, a1[ks], false, b,
                                                              (short)0, acc, false, false);
            }
            const int   colb = nt * 16 + row;
            const float bias = b1r[nt];
            #pragma unroll
            for (int v = 0; v < 8; ++v) {
                float hv = acc[v] + bias;
                hv = hv > 0.f ? hv : 0.f;
                hb[(kb + v) * 256 + colb] = (__bf16)hv;   // C-layout -> row-major LDS
            }
        }

        asm volatile("s_wait_dscnt 0" ::: "memory");

        // ---- A fragments for GEMM2 from h tile (row-major LDS -> A layout) ----
        v16bf a2[8];
        #pragma unroll
        for (int ks = 0; ks < 8; ++ks) {
            const __bf16* hp = hb + row * 256 + ks * 32 + kb;
            v16bf a;
            #pragma unroll
            for (int i = 0; i < 8; ++i) { a[i] = hp[i]; a[8 + i] = hp[16 + i]; }
            a2[ks] = a;
        }

        asm volatile("s_wait_dscnt 0" ::: "memory");

        // ---- GEMM2: theta = h @ W2 + b2 (N padded to 112), store f32 tile (reuses h buf) ----
        #pragma unroll
        for (int nt2 = 0; nt2 < 7; ++nt2) {
            v8f acc = {};
            #pragma unroll
            for (int ks = 0; ks < 8; ++ks) {
                v16bf b = *(const v16bf*)&sW2[(nt2 * 8 + ks) * 512 + l * 16];
                acc = __builtin_amdgcn_wmma_f32_16x16x32_bf16(false, a2[ks], false, b,
                                                              (short)0, acc, false, false);
            }
            const int   colb = nt2 * 16 + row;
            const float bias = b2r[nt2];
            #pragma unroll
            for (int v = 0; v < 8; ++v)
                tb[(kb + v) * 112 + colb] = acc[v] + bias;
        }

        asm volatile("s_wait_dscnt 0" ::: "memory");

        // ---- RQS spline + log-prob ----
        // Row `row` is handled by the lane pair (row, row+16): low lane owns bins
        // [0,16), high lane owns bins [16,32); combined via shfl_xor(16).
        const float* trow = tb + row * 112;
        const int    kOff = khalf * 16;
        const float  xv = x[(r0 + row) * 4 + 3];
        const float  xs = fminf(fmaxf(xv, -10.f), 10.f);

        // pass 1: max over this lane's half-range, combine to full max
        float mw = -1e30f, mh = -1e30f;
        #pragma unroll
        for (int k = 0; k < 16; ++k) {
            mw = fmaxf(mw, trow[kOff + k]);
            mh = fmaxf(mh, trow[32 + kOff + k]);
        }
        mw = fmaxf(mw, __shfl_xor(mw, 16, 32));
        mh = fmaxf(mh, __shfl_xor(mh, 16, 32));

        // pass 2: partial exp-sums; partner's partial is also this half's prefix
        float swp = 0.f, shp = 0.f;
        #pragma unroll
        for (int k = 0; k < 16; ++k) {
            swp += __expf(trow[kOff + k] - mw);
            shp += __expf(trow[32 + kOff + k] - mh);
        }
        const float swo = __shfl_xor(swp, 16, 32);
        const float sho = __shfl_xor(shp, 16, 32);
        const float rw = 0.99968f / (swp + swo);   // (1 - NB*MIN_W)/sum_exp
        const float rh = 0.99968f / (shp + sho);
        // cumulative start for this half (high half adds low half's prefix)
        const float pre = (float)khalf;
        float cw = -10.f + pre * 20.f * (16.f * 1e-5f + rw * swo);
        float ch = -10.f + pre * 20.f * (16.f * 1e-5f + rh * sho);

        // pass 3: cumsum over this half's 16 bins, capture last bin with cumw <= xs
        int   bin = 0, found = 0;
        float in_cw = -10.f, in_w = 20.f, in_ch = -10.f, in_h = 20.f;
        #pragma unroll
        for (int kk = 0; kk < 16; ++kk) {
            const int   k   = kOff + kk;
            const float wk  = 1e-5f + rw * __expf(trow[k] - mw);
            const float hk  = 1e-5f + rh * __expf(trow[32 + k] - mh);
            const float cwn = cw + 20.f * wk;
            const float chn = ch + 20.f * hk;
            const float cwh = (k == 31) ? 10.f : cwn;   // cumw[32] forced to +BOUND
            const float chh = (k == 31) ? 10.f : chn;
            if (xs >= cw) {   // last k with cumw[k] <= xs (== reference bin rule)
                found = 1; bin = k;
                in_cw = cw; in_w = cwh - cw; in_ch = ch; in_h = chh - ch;
            }
            cw = cwn; ch = chn;
        }
        // combine halves: high half wins iff it found any bin (i.e. xs >= cumw[16])
        {
            const int   f_hi  = __shfl_xor(found, 16, 32);
            const int   b_hi  = __shfl_xor(bin, 16, 32);
            const float cw_hi = __shfl_xor(in_cw, 16, 32);
            const float w_hi  = __shfl_xor(in_w, 16, 32);
            const float ch_hi = __shfl_xor(in_ch, 16, 32);
            const float h_hi  = __shfl_xor(in_h, 16, 32);
            const bool useHi = (khalf == 0) && f_hi;   // only low lanes produce output
            if (useHi) { bin = b_hi; in_cw = cw_hi; in_w = w_hi; in_ch = ch_hi; in_h = h_hi; }
        }

        const float d0 = 1e-5f + softplusf(trow[64 + bin]);
        const float d1 = 1e-5f + softplusf(trow[64 + bin + 1]);
        const float delta = in_h / in_w;
        const float u   = (xs - in_cw) / in_w;
        const float uo  = u * (1.f - u);
        const float num = in_h * (delta * u * u + d0 * uo);
        const float den = delta + (d0 + d1 - 2.f * delta) * uo;
        const float zin = in_ch + num / den;
        const float omu = 1.f - u;
        const float dnm = delta * delta * (d1 * u * u + 2.f * delta * uo + d0 * omu * omu);
        const float lad = __logf(dnm) - 2.f * __logf(den);

        const bool  inside = (xv >= -10.f) && (xv <= 10.f);
        const float z   = inside ? zin : xv;
        const float jac = inside ? lad : 0.f;
        float prob = -0.91893853320467274f - 0.5f * z * z + jac;

        prob = (l < 16) ? prob : 0.f;
        #pragma unroll
        for (int off = 16; off > 0; off >>= 1)
            prob += __shfl_xor(prob, off, 32);
        if (l == 0) out[t] = prob;
    }
}

extern "C" void kernel_launch(void* const* d_in, const int* in_sizes, int n_in,
                              void* d_out, int out_size, void* d_ws, size_t ws_size,
                              hipStream_t stream) {
    (void)in_sizes; (void)n_in; (void)out_size; (void)d_ws; (void)ws_size;
    const float* x  = (const float*)d_in[0];
    const float* c  = (const float*)d_in[1];
    const float* W1 = (const float*)d_in[2];
    const float* b1 = (const float*)d_in[3];
    const float* W2 = (const float*)d_in[4];
    const float* b2 = (const float*)d_in[5];
    float* out = (float*)d_out;

    const int blocks = NUM_TILES / (WAVES_PER_BLOCK * TILES_PER_WAVE); // 2048
    flow_fused_kernel<<<blocks, WAVES_PER_BLOCK * 32, 0, stream>>>(x, c, W1, b1, W2, b2, out);
}
